// GCNClassifier_38860864094898
// MI455X (gfx1250) — matrix operand order repaired and checked by reading, not compiled, not run on hardware
//
#include <hip/hip_runtime.h>
#include <hip/hip_bf16.h>

typedef __attribute__((ext_vector_type(16))) __bf16 v16bf;
typedef __attribute__((ext_vector_type(8)))  __bf16 v8bf;
typedef __attribute__((ext_vector_type(8)))  float  v8f;

static constexpr int kB = 512;
static constexpr int kS = 128;
static constexpr int kIN = 360;
static constexpr int kHID = 300;
static constexpr int kDDEP = 100;
static constexpr int kNCLS = 42;

#define BM 64
#define BN 128
#define BK 32
#define LDT 40   // LDS row stride in halfs (80B: 16B-aligned, bank-skewed)

// ---------------------------------------------------------------------------
// Generic batched GEMM: C = epilogue(A(f32,[M,K]) x B(f32,[K,N]))
// fp32 operands converted to bf16 during staging; f32 accumulation via
// v_wmma_f32_16x16x32_bf16. Double-buffered LDS software pipeline.
// loadA/loadB are pure raw loads from clamped (always-valid) addresses so the
// prefetched global loads stay in flight across compute; the 0/1 boundary
// mask is applied in storeT (recomputed from indices), so s_wait_loadcnt
// lands after the WMMAs. Epilogue: +Add, +bscale*bias[col], /(rowdiv[row]+1),
// relu.
// ---------------------------------------------------------------------------
__global__ __launch_bounds__(256)
void gemm_bf16_wmma(const float* __restrict__ A, long long sA, int lda,
                    const float* __restrict__ Bm, long long sB, int ldb,
                    const float* __restrict__ Add, long long sAdd, int ldadd, int hasAdd,
                    float* __restrict__ C, long long sC, int ldc,
                    int M, int N, int K,
                    const float* __restrict__ bias, float bscale,
                    const float* __restrict__ rowdiv, int divmode,
                    int relu)
{
  __shared__ alignas(16) __bf16 As[2][BM * LDT];
  __shared__ alignas(16) __bf16 Bs[2][BN * LDT];

  const int b = blockIdx.z;
  A  += (long long)b * sA;
  Bm += (long long)b * sB;
  C  += (long long)b * sC;
  if (hasAdd) Add += (long long)b * sAdd;

  const int m0 = blockIdx.x * BM;
  const int n0 = blockIdx.y * BN;
  const int tid  = threadIdx.x;
  const int lane = tid & 31;
  const int w    = tid >> 5;
  const int wm   = w & 3;   // 4 M-subtiles of 16
  const int wn   = w >> 2;  // 2 N-halves of 64

  // staging coordinates (per thread)
  const int aK  = tid & 31;    // A: column within BK
  const int aM  = tid >> 5;    // A: row base (+ j*8)
  const int bN  = tid & 127;   // B: column within BN
  const int bK0 = tid >> 7;    // B: k base (+ j*2)

  v8f acc[4];
#pragma unroll
  for (int t = 0; t < 4; t++)
#pragma unroll
    for (int r = 0; r < 8; r++) acc[t][r] = 0.0f;

  float ra[8], rb[16];

  // Pure raw loads from clamped, always-valid addresses. No consumers here:
  // the boundary mask is applied at commit time (storeT), so these loads can
  // remain outstanding across the WMMAs of the current tile.
  auto loadA = [&](int k0) {
#pragma unroll
    for (int j = 0; j < 8; j++) {
      int gm = m0 + aM + j * 8, gk = k0 + aK;
      int cm = gm < M ? gm : M - 1;
      int ck = gk < K ? gk : K - 1;
      ra[j] = A[(long long)cm * lda + ck];
    }
  };
  auto loadB = [&](int k0) {
#pragma unroll
    for (int j = 0; j < 16; j++) {
      int gk = k0 + bK0 + j * 2, gn = n0 + bN;
      int ck = gk < K ? gk : K - 1;
      int cn = gn < N ? gn : N - 1;
      rb[j] = Bm[(long long)ck * ldb + cn];
    }
  };
  // Commit regs -> LDS (bf16), applying the 0/1 boundary mask here.
  // (Multiply by 0 zeroes out-of-range elements; clamped loads are finite.)
  auto storeT = [&](int p, int k0) {
#pragma unroll
    for (int j = 0; j < 8; j++) {
      int gm = m0 + aM + j * 8, gk = k0 + aK;
      float msk = (gm < M && gk < K) ? 1.0f : 0.0f;
      As[p][(aM + j * 8) * LDT + aK] = (__bf16)(ra[j] * msk);
    }
#pragma unroll
    for (int j = 0; j < 16; j++) {
      int gk = k0 + bK0 + j * 2, gn = n0 + bN;
      float msk = (gk < K && gn < N) ? 1.0f : 0.0f;
      Bs[p][bN * LDT + bK0 + j * 2] = (__bf16)(rb[j] * msk);  // Bs[n][k]
    }
  };

  const int khalf = lane >> 4;
  const int arow  = (wm << 4) + (lane & 15);

  auto compute = [&](int p) {
    // A fragment: m = wm*16 + (lane&15); k(e) = (e>>3)*16 + khalf*8 + (e&7)
    const __bf16* ap = &As[p][arow * LDT + (khalf << 3)];
    v8bf alo = *(const v8bf*)ap;
    v8bf ahi = *(const v8bf*)(ap + 16);
    v8bf blo[4], bhi[4];
#pragma unroll
    for (int t = 0; t < 4; t++) {
      // B fragment: n = wn*64 + t*16 + (lane&15); k(e) = (e>>3)*16 + khalf*8 + (e&7)
      const __bf16* bp = &Bs[p][((wn << 6) + (t << 4) + (lane & 15)) * LDT + (khalf << 3)];
      blo[t] = *(const v8bf*)bp;
      bhi[t] = *(const v8bf*)(bp + 16);
    }
    v16bf afrag;
#pragma unroll
    for (int e = 0; e < 8; e++) { afrag[e] = alo[e]; afrag[e + 8] = ahi[e]; }
#pragma unroll
    for (int t = 0; t < 4; t++) {
      v16bf bfrag;
#pragma unroll
      for (int e = 0; e < 8; e++) { bfrag[e] = blo[t][e]; bfrag[e + 8] = bhi[t][e]; }
      acc[t] = __builtin_amdgcn_wmma_f32_16x16x32_bf16(
          false, afrag, false, bfrag, (short)0, acc[t], false, false);
    }
  };

  const int nk = (K + BK - 1) / BK;
  loadA(0); loadB(0); storeT(0, 0);
  __syncthreads();
  for (int kt = 0; kt < nk; kt++) {
    const int cur = kt & 1;
    const bool more = (kt + 1) < nk;
    const int knext = (kt + 1) * BK;
    if (more) { loadA(knext); loadB(knext); }   // prefetch to regs (in flight)
    compute(cur);                               // 4 WMMAs hide the HBM latency
    if (more) storeT(cur ^ 1, knext);           // wait + convert + commit
    __syncthreads();
  }

  // Epilogue: C/D layout row = (lane>>4)*8 + r, col = lane&15
#pragma unroll
  for (int t = 0; t < 4; t++) {
    int col = n0 + (wn << 6) + (t << 4) + (lane & 15);
    if (col >= N) continue;
    int rbase = m0 + (wm << 4) + ((lane >> 4) << 3);
#pragma unroll
    for (int r = 0; r < 8; r++) {
      int row = rbase + r;
      if (row >= M) break;
      float v = acc[t][r];
      if (hasAdd)  v += Add[(long long)row * ldadd + col];
      if (bias)    v += bscale * bias[col];
      if (divmode) v /= (rowdiv[row] + 1.0f);
      if (relu)    v = v > 0.0f ? v : 0.0f;
      C[(long long)row * ldc + col] = v;
    }
  }
}

// ---------------------------------------------------------------------------
// Degree sums: outdeg[b,s] = sum_t adj[b,s,t]; indeg[b,t] = sum_s adj[b,s,t]
// ---------------------------------------------------------------------------
__global__ __launch_bounds__(128)
void degrees_kernel(const float* __restrict__ adj, float* __restrict__ outdeg,
                    float* __restrict__ indeg)
{
  int b = blockIdx.x, t = threadIdx.x;
  const float* ab = adj + (long long)b * kS * kS;
  float so = 0.0f, si = 0.0f;
  for (int j = 0; j < kS; j++) {
    so += ab[t * kS + j];
    si += ab[j * kS + t];
  }
  outdeg[b * kS + t] = so;
  indeg[b * kS + t]  = si;
}

// ---------------------------------------------------------------------------
// Embedding gather: h0[b,s,:] = [emb_w[words]|pos_w[pos]|ner_w[ner]]
// ---------------------------------------------------------------------------
__global__ __launch_bounds__(128)
void gather_kernel(const int* __restrict__ words, const int* __restrict__ pos,
                   const int* __restrict__ ner,
                   const float* __restrict__ emb_w, const float* __restrict__ pos_w,
                   const float* __restrict__ ner_w, float* __restrict__ h0)
{
  long long i = blockIdx.x;   // b*S + s
  int wid = words[i], pid = pos[i], nid = ner[i];
  float* out = h0 + i * kIN;
  for (int d = threadIdx.x; d < kIN; d += blockDim.x) {
    float v;
    if (d < 300)      v = emb_w[(long long)wid * 300 + d];
    else if (d < 330) v = pos_w[pid * 30 + (d - 300)];
    else              v = ner_w[nid * 30 + (d - 330)];
    out[d] = v;
  }
}

// ---------------------------------------------------------------------------
// query[b,:] = masked max over s of h0 (mask: outdeg+indeg == 0 -> -1e12)
// ---------------------------------------------------------------------------
__global__ __launch_bounds__(128)
void query_pool_kernel(const float* __restrict__ h0, const float* __restrict__ outdeg,
                       const float* __restrict__ indeg, float* __restrict__ query)
{
  int b = blockIdx.x;
  for (int d = threadIdx.x; d < kIN; d += blockDim.x) {
    float m = -3.0e38f;
    for (int s = 0; s < kS; s++) {
      bool pm = (outdeg[b * kS + s] + indeg[b * kS + s]) == 0.0f;
      float v = pm ? -1.0e12f : h0[((long long)b * kS + s) * kIN + d];
      m = fmaxf(m, v);
    }
    query[b * kIN + d] = m;
  }
}

// ---------------------------------------------------------------------------
// scores/softmax: q2 = attn_w @ query[b]; score[s] = dep_w[deprel] . q2,
// masked by indeg==0 -> -1e30; softmax over s (block == S lanes).
// ---------------------------------------------------------------------------
__global__ __launch_bounds__(128)
void attn_kernel(const int* __restrict__ deprel, const float* __restrict__ dep_w,
                 const float* __restrict__ attn_w, const float* __restrict__ query,
                 const float* __restrict__ indeg, float* __restrict__ wts)
{
  __shared__ float qs[kIN];
  __shared__ float q2[kDDEP];
  __shared__ float red[kS];
  int b = blockIdx.x, tid = threadIdx.x;
  for (int e = tid; e < kIN; e += 128) qs[e] = query[b * kIN + e];
  __syncthreads();
  for (int d = tid; d < kDDEP; d += 128) {
    float s = 0.0f;
    const float* ar = attn_w + d * kIN;
    for (int e = 0; e < kIN; e++) s += ar[e] * qs[e];
    q2[d] = s;
  }
  __syncthreads();
  int s = tid;
  float sc;
  if (indeg[b * kS + s] == 0.0f) {
    sc = -1.0e30f;
  } else {
    const float* de = dep_w + (long long)deprel[b * kS + s] * kDDEP;
    float acc = 0.0f;
    for (int d = 0; d < kDDEP; d++) acc += de[d] * q2[d];
    sc = acc;
  }
  red[tid] = sc; __syncthreads();
  for (int o = 64; o > 0; o >>= 1) {
    if (tid < o) red[tid] = fmaxf(red[tid], red[tid + o]);
    __syncthreads();
  }
  float mx = red[0]; __syncthreads();
  float ex = __expf(sc - mx);
  red[tid] = ex; __syncthreads();
  for (int o = 64; o > 0; o >>= 1) {
    if (tid < o) red[tid] += red[tid + o];
    __syncthreads();
  }
  wts[b * kS + s] = ex / red[0];
}

// ---------------------------------------------------------------------------
// a[b,s,t] = adj[b,s,t]*w[b,s] + adj[b,t,s]*w[b,t]
// ---------------------------------------------------------------------------
__global__ __launch_bounds__(128)
void build_a_kernel(const float* __restrict__ adj, const float* __restrict__ wts,
                    float* __restrict__ a)
{
  int b = blockIdx.y, s = blockIdx.x, t = threadIdx.x;
  const float* ab = adj + (long long)b * kS * kS;
  float v = ab[s * kS + t] * wts[b * kS + s] + ab[t * kS + s] * wts[b * kS + t];
  a[((long long)b * kS + s) * kS + t] = v;
}

// ---------------------------------------------------------------------------
// Triple masked-max pool of final h; writes MLP input P[b,900] and h_out.
// ---------------------------------------------------------------------------
__global__ __launch_bounds__(128)
void pool3_kernel(const float* __restrict__ h2, const float* __restrict__ outdeg,
                  const float* __restrict__ indeg, const int* __restrict__ subj_pos,
                  const int* __restrict__ obj_pos, float* __restrict__ P,
                  float* __restrict__ hout)
{
  int b = blockIdx.x;
  for (int d = threadIdx.x; d < kHID; d += blockDim.x) {
    float m0 = -3.0e38f, m1 = -3.0e38f, m2 = -3.0e38f;
    for (int s = 0; s < kS; s++) {
      float h = h2[((long long)b * kS + s) * kHID + d];
      bool pm = (outdeg[b * kS + s] + indeg[b * kS + s]) == 0.0f;
      bool sm = subj_pos[b * kS + s] != 0;
      bool om = obj_pos[b * kS + s] != 0;
      m0 = fmaxf(m0, pm ? -1.0e12f : h);
      m1 = fmaxf(m1, sm ? -1.0e12f : h);
      m2 = fmaxf(m2, om ? -1.0e12f : h);
    }
    P[(long long)b * 900 + d]       = m0;
    P[(long long)b * 900 + 300 + d] = m1;
    P[(long long)b * 900 + 600 + d] = m2;
    hout[(long long)b * kHID + d]   = m0;
  }
}

// ---------------------------------------------------------------------------
extern "C" void kernel_launch(void* const* d_in, const int* in_sizes, int n_in,
                              void* d_out, int out_size, void* d_ws, size_t ws_size,
                              hipStream_t stream)
{
  const int*   words  = (const int*)d_in[0];
  const int*   pos    = (const int*)d_in[1];
  const int*   ner    = (const int*)d_in[2];
  const int*   deprel = (const int*)d_in[3];
  const int*   subj   = (const int*)d_in[4];
  const int*   obj    = (const int*)d_in[5];
  const float* adj    = (const float*)d_in[6];
  const float* emb_w  = (const float*)d_in[7];
  const float* pos_w  = (const float*)d_in[8];
  const float* ner_w  = (const float*)d_in[9];
  const float* dep_w  = (const float*)d_in[10];
  const float* attn_w = (const float*)d_in[11];
  const float* W0     = (const float*)d_in[12];
  const float* b0     = (const float*)d_in[13];
  const float* W1     = (const float*)d_in[14];
  const float* b1     = (const float*)d_in[15];
  const float* mlp0_w = (const float*)d_in[16];
  const float* mlp0_b = (const float*)d_in[17];
  const float* mlp1_w = (const float*)d_in[18];
  const float* mlp1_b = (const float*)d_in[19];
  const float* cls_w  = (const float*)d_in[20];
  const float* cls_b  = (const float*)d_in[21];

  float* out = (float*)d_out;          // [512*42 logits | 512*300 h_out]
  float* ws  = (float*)d_ws;

  size_t o = 0;
  float* h0   = ws + o; o += (size_t)kB * kS * kIN;   // h0, later M2 (layer 2)
  float* m2   = ws + o; o += (size_t)kB * kS * kIN;   // M2 (layer 1), later h2
  float* h1   = ws + o; o += (size_t)kB * kS * kHID;
  float* aw   = ws + o; o += (size_t)kB * kS * kS;
  float* outd = ws + o; o += (size_t)kB * kS;
  float* ind  = ws + o; o += (size_t)kB * kS;
  float* qry  = ws + o; o += (size_t)kB * kIN;
  float* wts  = ws + o; o += (size_t)kB * kS;
  float* P    = ws + o; o += (size_t)kB * 900;
  float* mo0  = ws + o; o += (size_t)kB * kHID;
  float* mo1  = ws + o; o += (size_t)kB * kHID;

  float* logits = out;
  float* hout   = out + (size_t)kB * kNCLS;

  degrees_kernel<<<kB, 128, 0, stream>>>(adj, outd, ind);
  gather_kernel<<<kB * kS, 128, 0, stream>>>(words, pos, ner, emb_w, pos_w, ner_w, h0);
  query_pool_kernel<<<kB, 128, 0, stream>>>(h0, outd, ind, qry);
  attn_kernel<<<kB, 128, 0, stream>>>(deprel, dep_w, attn_w, qry, ind, wts);
  build_a_kernel<<<dim3(kS, kB), 128, 0, stream>>>(adj, wts, aw);

  // Layer 1: M2 = a@h0 + h0  (batched, M=128, N=360, K=128)
  gemm_bf16_wmma<<<dim3(2, 3, kB), 256, 0, stream>>>(
      aw, (long long)kS * kS, kS,
      h0, (long long)kS * kIN, kIN,
      h0, (long long)kS * kIN, kIN, 1,
      m2, (long long)kS * kIN, kIN,
      kS, kIN, kS, nullptr, 0.0f, nullptr, 0, 0);
  // h1 = relu((M2 @ W0 + 2*b0) / (outdeg+1))   ([65536,360]@[360,300])
  gemm_bf16_wmma<<<dim3((kB * kS) / BM, 3, 1), 256, 0, stream>>>(
      m2, 0, kIN,
      W0, 0, kHID,
      nullptr, 0, 0, 0,
      h1, 0, kHID,
      kB * kS, kHID, kIN, b0, 2.0f, outd, 1, 1);
  // Layer 2: M2' = a@h1 + h1  (into h0 buffer)
  gemm_bf16_wmma<<<dim3(2, 3, kB), 256, 0, stream>>>(
      aw, (long long)kS * kS, kS,
      h1, (long long)kS * kHID, kHID,
      h1, (long long)kS * kHID, kHID, 1,
      h0, (long long)kS * kHID, kHID,
      kS, kHID, kS, nullptr, 0.0f, nullptr, 0, 0);
  // h2 = relu((M2' @ W1 + 2*b1) / (outdeg+1))  (into m2 buffer)
  gemm_bf16_wmma<<<dim3((kB * kS) / BM, 3, 1), 256, 0, stream>>>(
      h0, 0, kHID,
      W1, 0, kHID,
      nullptr, 0, 0, 0,
      m2, 0, kHID,
      kB * kS, kHID, kHID, b1, 2.0f, outd, 1, 1);

  pool3_kernel<<<kB, 128, 0, stream>>>(m2, outd, ind, subj, obj, P, hout);

  // MLP head
  gemm_bf16_wmma<<<dim3(kB / BM, 3, 1), 256, 0, stream>>>(
      P, 0, 900, mlp0_w, 0, kHID, nullptr, 0, 0, 0,
      mo0, 0, kHID, kB, kHID, 900, mlp0_b, 1.0f, nullptr, 0, 1);
  gemm_bf16_wmma<<<dim3(kB / BM, 3, 1), 256, 0, stream>>>(
      mo0, 0, kHID, mlp1_w, 0, kHID, nullptr, 0, 0, 0,
      mo1, 0, kHID, kB, kHID, kHID, mlp1_b, 1.0f, nullptr, 0, 1);
  gemm_bf16_wmma<<<dim3(kB / BM, 1, 1), 256, 0, stream>>>(
      mo1, 0, kHID, cls_w, 0, kNCLS, nullptr, 0, 0, 0,
      logits, 0, kNCLS, kB, kNCLS, kHID, cls_b, 1.0f, nullptr, 0, 0);
}
